// CausalDownsample_73632919322768
// MI455X (gfx1250) — matrix operand order repaired and checked by reading, not compiled, not run on hardware
//
#include <hip/hip_runtime.h>
#include <hip/hip_bf16.h>

typedef __attribute__((ext_vector_type(16))) _Float16 v16h;
typedef __attribute__((ext_vector_type(8)))  _Float16 v8h;
typedef __attribute__((ext_vector_type(8)))  float    v8f;

union F16x16 { v16h v; uint4 q[2]; };

#define DIMC 1024
#define NHEADS 8
#define HD 128
#define TFULL 4096
#define TQ2 1024

// low 32 bits of a generic pointer to __shared__ = LDS byte address
__device__ __forceinline__ unsigned lds_addr(const void* p) {
  return (unsigned)(unsigned long long)p;
}
// CDNA5 async global->LDS copy (VGLOBAL op 98), tracked by ASYNCcnt
__device__ __forceinline__ void async_ld_b128(unsigned lds_off, const _Float16* g) {
  asm volatile("global_load_async_to_lds_b128 %0, %1, off"
               :: "v"(lds_off), "v"((unsigned long long)(size_t)g) : "memory");
}
__device__ __forceinline__ void wait_async0() {
  asm volatile("s_wait_asynccnt 0" ::: "memory");
}

// ---------------------------------------------------------------------------
// x f32 -> f16 (same TM layout, vectorized 8-wide)
// ---------------------------------------------------------------------------
__global__ __launch_bounds__(256)
void cvt_f16_kernel(const float* __restrict__ x, _Float16* __restrict__ y, long total) {
  long idx = ((long)blockIdx.x * 256 + threadIdx.x) * 8;
  if (idx >= total) return;
  union { v8h v; _Float16 e[8]; } pk;
#pragma unroll
  for (int j = 0; j < 8; ++j) pk.e[j] = (_Float16)x[idx + j];
  *(v8h*)(y + idx) = pk.v;
}

// ---------------------------------------------------------------------------
// Pack weights f32 [O][I][Kw] -> f16 A-fragment order:
// Wp[chunk][m][32], chunk = kw*(I/32)+i0/32, slot p = hi*16+e, k = p
// ---------------------------------------------------------------------------
__global__ __launch_bounds__(256)
void pack_w_kernel(const float* __restrict__ W, _Float16* __restrict__ Wp,
                   int O, int I, int Kw, long total) {
  long idx = (long)blockIdx.x * 256 + threadIdx.x;
  if (idx >= total) return;
  int p = (int)(idx & 31);
  long rest = idx >> 5;
  int m = (int)(rest % O);
  int chunk = (int)(rest / O);
  int cpk = I >> 5;
  int kw = chunk / cpk;
  int i0 = (chunk - kw * cpk) << 5;
  int hi = p >> 4, e = p & 15;
  int k = e + (hi << 4);
  Wp[idx] = (_Float16)W[((size_t)m * I + (i0 + k)) * Kw + kw];
}

// ---------------------------------------------------------------------------
// Causal-conv / linear GEMM via WMMA. Activations t-major [B][T][C] f16 so the
// B-fragment (reduction = input channel) is ONE contiguous v16h global load.
// A (packed weights) double-buffered through LDS via async global->LDS copies.
// Block 256 thr = 8 waves (2x4), tile 128(M) x 128(N); wave 64x32, 8 WMMA/chunk.
// MODE: 0 = TM f16 [B][T][O] (+opt residual TM), 1 = CM f16 [B][O][T], 2 = TM f32.
// ---------------------------------------------------------------------------
template<int KW, int STRIDE, int DIL, int PAD, int TIN, int TOUT, int RELU, int MODE>
__global__ __launch_bounds__(256)
void conv_gemm_kernel(const _Float16* __restrict__ Wp, const float* __restrict__ bias,
                      const _Float16* __restrict__ act, const _Float16* res,
                      _Float16* out_h, float* out_f) {
  constexpr int IC = DIMC, O = DIMC;
  constexpr int CPK = IC / 32;
  constexpr int NCH = KW * CPK;
  const int wave = threadIdx.x >> 5;
  const int lane = threadIdx.x & 31;
  const int wm = wave >> 2, wn = wave & 3;     // 2 x 4 waves
  const int b  = blockIdx.z;
  const int mB = blockIdx.y * 128;
  const int nB = blockIdx.x * 128;
  const int m0w = wm * 64;                     // block-local
  const int n0w = wn * 32;
  const int nl = lane & 15, hi = lane >> 4;
  const int tid = threadIdx.x;

  __shared__ __align__(32) _Float16 abuf[2][128 * 32];  // double-buffered A tile (8KB each)

  // prefill buffer 0 with chunk 0 (each thread 32B = 2 async b128)
  {
    const _Float16* g = Wp + ((size_t)mB) * 32 + tid * 16;
    async_ld_b128(lds_addr(&abuf[0][tid * 16]), g);
    async_ld_b128(lds_addr(&abuf[0][tid * 16 + 8]), g + 8);
  }
  wait_async0();
  __syncthreads();

  v8f c[4][2] = {};

  for (int ch = 0; ch < NCH; ++ch) {
    const int cur = ch & 1;
    if (ch + 1 < NCH) {  // async-fill next chunk while computing this one
      const _Float16* g = Wp + ((size_t)(ch + 1) * O + mB) * 32 + tid * 16;
      async_ld_b128(lds_addr(&abuf[cur ^ 1][tid * 16]), g);
      async_ld_b128(lds_addr(&abuf[cur ^ 1][tid * 16 + 8]), g + 8);
    }
    const int kw = ch / CPK;
    const int i0 = (ch - kw * CPK) * 32;

    // A fragments from LDS (contiguous 32B -> ds_load_b128 pair)
    v16h a[4];
#pragma unroll
    for (int mi = 0; mi < 4; ++mi)
      a[mi] = *(const v16h*)&abuf[cur][(m0w + mi * 16 + nl) * 32 + (hi << 4)];

    // B fragments straight from global (contiguous along input channel)
    v16h bf[2];
#pragma unroll
    for (int ni = 0; ni < 2; ++ni) {
      const int t = nB + n0w + ni * 16 + nl;
      const int tin = t * STRIDE + kw * DIL - PAD;
      v16h bv = {};
      if (tin >= 0 && tin < TIN)
        bv = *(const v16h*)(act + ((size_t)b * TIN + tin) * IC + i0 + (hi << 4));
      if (RELU) {
#pragma unroll
        for (int e = 0; e < 16; ++e)
          if ((float)bv[e] < 0.f) bv[e] = (_Float16)0;
      }
      bf[ni] = bv;
    }

#pragma unroll
    for (int mi = 0; mi < 4; ++mi)
#pragma unroll
      for (int ni = 0; ni < 2; ++ni)
        c[mi][ni] = __builtin_amdgcn_wmma_f32_16x16x32_f16(
            false, a[mi], false, bf[ni], (short)0, c[mi][ni], false, false);

    wait_async0();     // next buffer landed
    __syncthreads();   // all waves done reading cur + see the new fill
  }

  // epilogue (C layout: m = j + 8*hi, n = lane&15)
#pragma unroll
  for (int mi = 0; mi < 4; ++mi)
#pragma unroll
    for (int ni = 0; ni < 2; ++ni) {
      const int t = nB + n0w + ni * 16 + nl;
      const int mbase = mB + m0w + mi * 16 + (hi << 3);
      if (MODE == 1) {                 // CM f16 (V projection)
#pragma unroll
        for (int j = 0; j < 8; ++j)
          out_h[((size_t)b * O + mbase + j) * TOUT + t] =
              (_Float16)(c[mi][ni][j] + bias[mbase + j]);
      } else if (MODE == 0) {          // TM f16, optional TM residual, vector store
        union { v8h v; _Float16 e[8]; } pk;
        if (res) {
          v8h r = *(const v8h*)&res[((size_t)b * TOUT + t) * O + mbase];
#pragma unroll
          for (int j = 0; j < 8; ++j)
            pk.e[j] = (_Float16)(c[mi][ni][j] + bias[mbase + j] + (float)r[j]);
        } else {
#pragma unroll
          for (int j = 0; j < 8; ++j)
            pk.e[j] = (_Float16)(c[mi][ni][j] + bias[mbase + j]);
        }
        *(v8h*)&out_h[((size_t)b * TOUT + t) * O + mbase] = pk.v;
      } else {                         // TM f32 vector store (final output)
        v8f pv;
#pragma unroll
        for (int j = 0; j < 8; ++j) pv[j] = c[mi][ni][j] + bias[mbase + j];
        *(v8f*)&out_f[((size_t)b * TOUT + t) * O + mbase] = pv;
      }
    }
}

// ---------------------------------------------------------------------------
// Fused cross-causal flash attention.
// q TM [B][TQ2][C], k TM [B][T][C], v CM [B][C][T] -> o TM [B][TQ2][C]
// mask: tk < (tq+1)*4 (also truncates the K loop). Wave owns 16 q rows.
// ---------------------------------------------------------------------------
__global__ __launch_bounds__(256)
void flash_attn_kernel(const _Float16* __restrict__ q,
                       const _Float16* __restrict__ k,
                       const _Float16* __restrict__ v,
                       _Float16* __restrict__ o) {
  const int lane = threadIdx.x & 31;
  const int wave = threadIdx.x >> 5;
  const int h = blockIdx.y, b = blockIdx.z;
  const int q0 = blockIdx.x * 128 + wave * 16;
  const int nl = lane & 15, hi = lane >> 4;
  const float scale = 0.08838834764831845f;      // 1/sqrt(128)

  __shared__ _Float16 plds[8][16 * 32];
  _Float16* my = &plds[wave][0];

  // Q fragments (A 16x32): two contiguous 8-half runs per lane -> 2x b128
  F16x16 aq[4];
  {
    const _Float16* qb = q + ((size_t)b * TQ2 + q0 + nl) * DIMC + h * HD;
#pragma unroll
    for (int dc = 0; dc < 4; ++dc) {
      aq[dc].q[0] = *(const uint4*)(qb + dc * 32 + (hi << 3));
      aq[dc].q[1] = *(const uint4*)(qb + dc * 32 + 16 + (hi << 3));
    }
  }

  float mrow[8], lrow[8];
  v8f oacc[8] = {};
#pragma unroll
  for (int j = 0; j < 8; ++j) { mrow[j] = -1e30f; lrow[j] = 0.f; }

  const int kmax = (q0 + 16) * 4;
  for (int kb = 0; kb < kmax; kb += 32) {
    // ---- S = Q^T K ----
    v8f s[2] = {};
#pragma unroll
    for (int ni = 0; ni < 2; ++ni) {
      const _Float16* kbase = k + ((size_t)b * TFULL + kb + ni * 16 + nl) * DIMC + h * HD;
      __builtin_prefetch(kbase + 32 * DIMC, 0, 1);   // next K block
#pragma unroll
      for (int dc = 0; dc < 4; ++dc) {
        F16x16 bk;
        bk.q[0] = *(const uint4*)(kbase + dc * 32 + (hi << 3));
        bk.q[1] = *(const uint4*)(kbase + dc * 32 + 16 + (hi << 3));
        s[ni] = __builtin_amdgcn_wmma_f32_16x16x32_f16(
            false, aq[dc].v, false, bk.v, (short)0, s[ni], false, false);
      }
    }

    // ---- scale + causal mask + online softmax ----
    float nm[8];
#pragma unroll
    for (int j = 0; j < 8; ++j) {
      int tq = q0 + j + (hi << 3);
      int lim = (tq + 1) * 4;
#pragma unroll
      for (int ni = 0; ni < 2; ++ni) {
        int tk = kb + ni * 16 + nl;
        float sv = s[ni][j] * scale;
        if (tk >= lim) sv = -1e30f;
        s[ni][j] = sv;
      }
      float mx = fmaxf(s[0][j], s[1][j]);
#pragma unroll
      for (int d = 1; d < 16; d <<= 1) mx = fmaxf(mx, __shfl_xor(mx, d, 32));
      nm[j] = fmaxf(mrow[j], mx);
    }
#pragma unroll
    for (int j = 0; j < 8; ++j) {
      float p0 = __expf(s[0][j] - nm[j]);
      float p1 = __expf(s[1][j] - nm[j]);
      s[0][j] = p0; s[1][j] = p1;
      float rs = p0 + p1;
#pragma unroll
      for (int d = 1; d < 16; d <<= 1) rs += __shfl_xor(rs, d, 32);
      float alpha = __expf(mrow[j] - nm[j]);
      lrow[j] = lrow[j] * alpha + rs;
      mrow[j] = nm[j];
#pragma unroll
      for (int dt = 0; dt < 8; ++dt) oacc[dt][j] *= alpha;
    }

    // ---- P: C-layout -> A-layout via per-wave LDS (DS in-order per wave) ----
#pragma unroll
    for (int ni = 0; ni < 2; ++ni)
#pragma unroll
      for (int j = 0; j < 8; ++j)
        my[(j + (hi << 3)) * 32 + ni * 16 + nl] = (_Float16)s[ni][j];
    __asm__ volatile("" ::: "memory");
    F16x16 pa;
    pa.q[0] = *(const uint4*)&my[nl * 32 + (hi << 3)];
    pa.q[1] = *(const uint4*)&my[nl * 32 + 16 + (hi << 3)];

    // ---- O += P V  (V CM -> contiguous along tk) ----
#pragma unroll
    for (int dt = 0; dt < 8; ++dt) {
      const _Float16* vb = v + ((size_t)b * DIMC + h * HD + dt * 16 + nl) * TFULL + kb;
      F16x16 bv;
      bv.q[0] = *(const uint4*)(vb + (hi << 3));
      bv.q[1] = *(const uint4*)(vb + 16 + (hi << 3));
      oacc[dt] = __builtin_amdgcn_wmma_f32_16x16x32_f16(
          false, pa.v, false, bv.v, (short)0, oacc[dt], false, false);
    }
  }

  // ---- normalize, store TM [B][TQ2][C] ----
#pragma unroll
  for (int j = 0; j < 8; ++j) {
    float inv = 1.f / lrow[j];
    int tq = q0 + j + (hi << 3);
#pragma unroll
    for (int dt = 0; dt < 8; ++dt) {
      int cch = h * HD + dt * 16 + nl;
      o[((size_t)b * TQ2 + tq) * DIMC + cch] = (_Float16)(oacc[dt][j] * inv);
    }
  }
}

// ---------------------------------------------------------------------------
extern "C" void kernel_launch(void* const* d_in, const int* in_sizes, int n_in,
                              void* d_out, int out_size, void* d_ws, size_t ws_size,
                              hipStream_t stream) {
  const float* x      = (const float*)d_in[0];
  const float* dw0    = (const float*)d_in[1];
  const float* db0    = (const float*)d_in[2];
  const float* rw1_0  = (const float*)d_in[3];
  const float* rb1_0  = (const float*)d_in[4];
  const float* rw2_0  = (const float*)d_in[5];
  const float* rb2_0  = (const float*)d_in[6];
  const float* dw1    = (const float*)d_in[7];
  const float* db1    = (const float*)d_in[8];
  const float* rw1_1  = (const float*)d_in[9];
  const float* rb1_1  = (const float*)d_in[10];
  const float* rw2_1  = (const float*)d_in[11];
  const float* rb2_1  = (const float*)d_in[12];
  const float* inpw   = (const float*)d_in[13];
  const float* inpb   = (const float*)d_in[14];
  const float* outw   = (const float*)d_in[15];
  const float* outb   = (const float*)d_in[16];

  const int Bn = 2, T = TFULL, C = DIMC;
  const int T1 = T / 2, T2 = T / 4;

  char* ws = (char*)d_ws;
  auto alloc_h = [&](size_t n) { _Float16* p = (_Float16*)ws; ws += n * 2; return p; };

  // activations (t-major [B][T][C] unless noted)
  _Float16* xh   = alloc_h((size_t)Bn * T  * C);   // x as f16 TM
  _Float16* s1a  = alloc_h((size_t)Bn * T1 * C);
  _Float16* s1h  = alloc_h((size_t)Bn * T1 * C);
  _Float16* s2a  = alloc_h((size_t)Bn * T2 * C);
  _Float16* s2h  = alloc_h((size_t)Bn * T2 * C);
  _Float16* qbuf = alloc_h((size_t)Bn * T2 * C);
  _Float16* kbuf = alloc_h((size_t)Bn * T  * C);
  _Float16* vbuf = alloc_h((size_t)Bn * C  * T);   // CM
  _Float16* obuf = alloc_h((size_t)Bn * T2 * C);
  // packed f16 weights
  const size_t W1 = (size_t)C * C;
  _Float16* wp_dw0  = alloc_h(W1 * 4);
  _Float16* wp_dw1  = alloc_h(W1 * 4);
  _Float16* wp_r1a  = alloc_h(W1 * 3 * 3);
  _Float16* wp_r1b  = alloc_h(W1 * 3);
  _Float16* wp_r2a  = alloc_h(W1 * 3 * 3);
  _Float16* wp_r2b  = alloc_h(W1 * 3);
  _Float16* wp_q    = alloc_h(W1);
  _Float16* wp_k    = alloc_h(W1);
  _Float16* wp_v    = alloc_h(W1);
  _Float16* wp_o    = alloc_h(W1);

  dim3 blk(256);
  auto pack = [&](const float* W, _Float16* Wp, int O, int I, int Kw) {
    long total = (long)O * I * Kw;
    pack_w_kernel<<<(unsigned)((total + 255) / 256), blk, 0, stream>>>(W, Wp, O, I, Kw, total);
  };

  // ---- weight packing ----
  pack(dw0, wp_dw0, C, C, 4);
  pack(dw1, wp_dw1, C, C, 4);
  for (int j = 0; j < 3; ++j) {
    pack(rw1_0 + (size_t)j * C * C * 3, wp_r1a + (size_t)j * C * C * 3, C, C, 3);
    pack(rw2_0 + (size_t)j * C * C,     wp_r1b + (size_t)j * C * C,     C, C, 1);
    pack(rw1_1 + (size_t)j * C * C * 3, wp_r2a + (size_t)j * C * C * 3, C, C, 3);
    pack(rw2_1 + (size_t)j * C * C,     wp_r2b + (size_t)j * C * C,     C, C, 1);
  }
  pack(inpw,          wp_q, C, C, 1);
  pack(inpw + W1,     wp_k, C, C, 1);
  pack(inpw + 2 * W1, wp_v, C, C, 1);
  pack(outw,          wp_o, C, C, 1);

  // ---- x f32 -> f16 (TM, no transpose needed) ----
  {
    long total = (long)Bn * T * C;
    cvt_f16_kernel<<<(unsigned)((total / 8 + 255) / 256), blk, 0, stream>>>(x, xh, total);
  }

  const dim3 g4096(T  / 128, C / 128, Bn);
  const dim3 g2048(T1 / 128, C / 128, Bn);
  const dim3 g1024(T2 / 128, C / 128, Bn);

  // ---- stage 1: downsample conv(k4,s2,pad3) + 3 causal resnet blocks ----
  conv_gemm_kernel<4,2,1, 3,4096,2048,0,0><<<g2048, blk, 0, stream>>>(wp_dw0, db0, xh, nullptr, s1a, nullptr);
  conv_gemm_kernel<3,1,9,18,2048,2048,1,0><<<g2048, blk, 0, stream>>>(wp_r1a,          rb1_0,         s1a, nullptr, s1h, nullptr);
  conv_gemm_kernel<1,1,1, 0,2048,2048,1,0><<<g2048, blk, 0, stream>>>(wp_r1b,          rb2_0,         s1h, s1a,    s1a, nullptr);
  conv_gemm_kernel<3,1,3, 6,2048,2048,1,0><<<g2048, blk, 0, stream>>>(wp_r1a + 3 * W1, rb1_0 + C,     s1a, nullptr, s1h, nullptr);
  conv_gemm_kernel<1,1,1, 0,2048,2048,1,0><<<g2048, blk, 0, stream>>>(wp_r1b + W1,     rb2_0 + C,     s1h, s1a,    s1a, nullptr);
  conv_gemm_kernel<3,1,1, 2,2048,2048,1,0><<<g2048, blk, 0, stream>>>(wp_r1a + 6 * W1, rb1_0 + 2 * C, s1a, nullptr, s1h, nullptr);
  conv_gemm_kernel<1,1,1, 0,2048,2048,1,0><<<g2048, blk, 0, stream>>>(wp_r1b + 2 * W1, rb2_0 + 2 * C, s1h, s1a,    s1a, nullptr);

  // ---- stage 2 ----
  conv_gemm_kernel<4,2,1, 3,2048,1024,0,0><<<g1024, blk, 0, stream>>>(wp_dw1, db1, s1a, nullptr, s2a, nullptr);
  conv_gemm_kernel<3,1,9,18,1024,1024,1,0><<<g1024, blk, 0, stream>>>(wp_r2a,          rb1_1,         s2a, nullptr, s2h, nullptr);
  conv_gemm_kernel<1,1,1, 0,1024,1024,1,0><<<g1024, blk, 0, stream>>>(wp_r2b,          rb2_1,         s2h, s2a,    s2a, nullptr);
  conv_gemm_kernel<3,1,3, 6,1024,1024,1,0><<<g1024, blk, 0, stream>>>(wp_r2a + 3 * W1, rb1_1 + C,     s2a, nullptr, s2h, nullptr);
  conv_gemm_kernel<1,1,1, 0,1024,1024,1,0><<<g1024, blk, 0, stream>>>(wp_r2b + W1,     rb2_1 + C,     s2h, s2a,    s2a, nullptr);
  conv_gemm_kernel<3,1,1, 2,1024,1024,1,0><<<g1024, blk, 0, stream>>>(wp_r2a + 6 * W1, rb1_1 + 2 * C, s2a, nullptr, s2h, nullptr);
  conv_gemm_kernel<1,1,1, 0,1024,1024,1,0><<<g1024, blk, 0, stream>>>(wp_r2b + 2 * W1, rb2_1 + 2 * C, s2h, s2a,    s2a, nullptr);

  // ---- projections: q (TM), k (TM), v (CM) ----
  conv_gemm_kernel<1,1,1,0,1024,1024,0,0><<<g1024, blk, 0, stream>>>(wp_q, inpb,         s2a, nullptr, qbuf, nullptr);
  conv_gemm_kernel<1,1,1,0,4096,4096,0,0><<<g4096, blk, 0, stream>>>(wp_k, inpb + C,     xh,  nullptr, kbuf, nullptr);
  conv_gemm_kernel<1,1,1,0,4096,4096,0,1><<<g4096, blk, 0, stream>>>(wp_v, inpb + 2 * C, xh,  nullptr, vbuf, nullptr);

  // ---- fused cross-causal attention ----
  {
    dim3 ag(T2 / 128, NHEADS, Bn);
    flash_attn_kernel<<<ag, blk, 0, stream>>>(qbuf, kbuf, vbuf, obuf);
  }

  // ---- output projection -> d_out f32 [B][Tq][C] ----
  conv_gemm_kernel<1,1,1,0,1024,1024,0,2><<<g1024, blk, 0, stream>>>(wp_o, outb, obuf, nullptr, nullptr, (float*)d_out);
}
